// GPT2Block_20727512171020
// MI455X (gfx1250) — matrix operand order repaired and checked by reading, not compiled, not run on hardware
//
#include <hip/hip_runtime.h>
#include <math.h>

// ---------------- problem constants ----------------
#define DMODEL 1024
#define FFDIM  4096
#define NHEAD  16
#define HDIM   64
#define SEQ    2048
#define BATCH  2
#define NROWS  (BATCH * SEQ)   // 4096 token rows

// ---------------- WMMA types ----------------
typedef __attribute__((ext_vector_type(16))) __bf16 v16bf;
typedef __attribute__((ext_vector_type(8)))  float  v8f;

union Frag16 { v16bf v; uint4 q[2]; unsigned short s[16]; };
union FragC  { v8f  v; float f[8]; };

__device__ __forceinline__ unsigned short f2bf(float x) {
    unsigned u = __float_as_uint(x);
    u += 0x7FFFu + ((u >> 16) & 1u);     // round to nearest even
    return (unsigned short)(u >> 16);
}

__device__ __forceinline__ v8f wmma_bf16(const Frag16& a, const Frag16& b, v8f c) {
    // (neg_a, A, neg_b, B, c_mod, C, reuse_a, reuse_b)
    return __builtin_amdgcn_wmma_f32_16x16x32_bf16(false, a.v, false, b.v,
                                                   (short)0, c, false, false);
}

// ---------------- weight pre-pack: f32 [K][N] -> bf16 fragment-order ----------------
// Layout: dword index = (((t*KS + ks)*32 + lane)*8 + dw)
//   element pair (2*dw, 2*dw+1) of B-fragment for n-tile t, k-step ks, lane L:
//   B[ks*32 + (dw>>2)*16 + (L>>4)*8 + 2*(dw&3) + j, t*16 + (L&15)]
__global__ __launch_bounds__(256) void pack_b_kernel(const float* __restrict__ W,
                                                     unsigned* __restrict__ P,
                                                     int K, int N) {
    int flat = blockIdx.x * 256 + threadIdx.x;
    int total = (K * N) >> 1;
    if (flat >= total) return;
    int dw   = flat & 7;
    int L    = (flat >> 3) & 31;
    int rest = flat >> 8;
    int KS   = K >> 5;
    int ks   = rest % KS;
    int t    = rest / KS;
    int r0   = ks * 32 + (L >> 4) * 8 + (dw >> 2) * 16 + (dw & 3) * 2;
    int c    = t * 16 + (L & 15);
    unsigned lo = f2bf(W[(size_t)r0 * N + c]);
    unsigned hi = f2bf(W[(size_t)(r0 + 1) * N + c]);
    P[flat] = lo | (hi << 16);
}

__global__ __launch_bounds__(256) void concat3_kernel(const float* __restrict__ a,
                                                      const float* __restrict__ b,
                                                      const float* __restrict__ c,
                                                      float* __restrict__ o) {
    int i = blockIdx.x * 256 + threadIdx.x;     // 3072 total
    o[i] = (i < 1024) ? a[i] : (i < 2048 ? b[i - 1024] : c[i - 2048]);
}

// ---------------- LayerNorm: f32 row -> bf16 row ----------------
__global__ __launch_bounds__(256) void ln_bf16_kernel(const float* __restrict__ x,
                                                      const float* __restrict__ g,
                                                      const float* __restrict__ beta,
                                                      unsigned short* __restrict__ out) {
    __shared__ float red[16];
    const int row = blockIdx.x;
    const float* xr = x + (size_t)row * DMODEL;
    const int i0 = threadIdx.x * 4;
    float4 v = *(const float4*)(xr + i0);
    float s  = v.x + v.y + v.z + v.w;
    float ss = v.x * v.x + v.y * v.y + v.z * v.z + v.w * v.w;
    for (int off = 1; off < 32; off <<= 1) {
        s  += __shfl_xor(s, off);
        ss += __shfl_xor(ss, off);
    }
    const int lane = threadIdx.x & 31, wid = threadIdx.x >> 5;
    if (lane == 0) { red[wid] = s; red[8 + wid] = ss; }
    __syncthreads();
    s = 0.f; ss = 0.f;
    for (int i = 0; i < 8; i++) { s += red[i]; ss += red[8 + i]; }
    const float mean = s * (1.0f / DMODEL);
    const float var  = ss * (1.0f / DMODEL) - mean * mean;
    const float rstd = rsqrtf(var + 1e-5f);
    float4 gg = *(const float4*)(g + i0);
    float4 bb = *(const float4*)(beta + i0);
    float h0 = (v.x - mean) * rstd * gg.x + bb.x;
    float h1 = (v.y - mean) * rstd * gg.y + bb.y;
    float h2 = (v.z - mean) * rstd * gg.z + bb.z;
    float h3 = (v.w - mean) * rstd * gg.w + bb.w;
    uint2 p;
    p.x = (unsigned)f2bf(h0) | ((unsigned)f2bf(h1) << 16);
    p.y = (unsigned)f2bf(h2) | ((unsigned)f2bf(h3) << 16);
    *(uint2*)(out + (size_t)row * DMODEL + i0) = p;
}

// ---------------- bf16 WMMA GEMM ----------------
// C[M=4096, N] = A(bf16, row-major, lda) x Bpacked + bias, epilogues:
//   epi 0: store bf16(acc + bias)
//   epi 1: store bf16(gelu(acc + bias))     (exact erf gelu)
//   epi 2: store f32 (acc + bias + resid)
// Block: 256 thr = 8 waves, block tile 128x128; wave tile 64(M) x 32(N).
__global__ __launch_bounds__(256) void gemm_bf16_wmma(
        const unsigned short* __restrict__ A, int lda,
        const unsigned* __restrict__ Bp, int K,
        const float* __restrict__ bias,
        const float* __restrict__ resid,
        void* __restrict__ outp, int ldc, int epi) {
    __shared__ unsigned short As[128 * 40];   // 128 rows x 32 bf16, padded stride 40
    const int lane = threadIdx.x & 31;
    const int wid  = threadIdx.x >> 5;
    const int mw = wid & 1, nw = wid >> 1;
    const int l15 = lane & 15, hs = lane >> 4;
    const int mBase = blockIdx.y * 128;
    const int nBase = blockIdx.x * 128;
    const int KS = K >> 5;

    FragC acc[4][2];
    #pragma unroll
    for (int i = 0; i < 4; i++)
        #pragma unroll
        for (int j = 0; j < 2; j++)
            #pragma unroll
            for (int e = 0; e < 8; e++) acc[i][j].f[e] = 0.f;

    const int crow = threadIdx.x >> 1;   // coop A load: 128 rows, 2 thr/row
    const int cseg = threadIdx.x & 1;    // 16-element (32B) halves

    for (int ks = 0; ks < KS; ++ks) {
        __syncthreads();
        {
            const uint4* src = (const uint4*)(A + (size_t)(mBase + crow) * lda + ks * 32 + cseg * 16);
            uint4 a0 = src[0], a1 = src[1];
            uint4* dst = (uint4*)(&As[crow * 40 + cseg * 16]);
            dst[0] = a0; dst[1] = a1;
        }
        __syncthreads();

        Frag16 bf[2];
        #pragma unroll
        for (int f = 0; f < 2; ++f) {
            const unsigned* bp = Bp + (((size_t)((nBase >> 4) + nw * 2 + f) * KS + ks) * 32 + lane) * 8;
            bf[f].q[0] = ((const uint4*)bp)[0];
            bf[f].q[1] = ((const uint4*)bp)[1];
        }
        #pragma unroll
        for (int sm = 0; sm < 4; ++sm) {
            Frag16 af;
            const unsigned short* ap = &As[(mw * 64 + sm * 16 + l15) * 40 + hs * 8];
            af.q[0] = *(const uint4*)(ap);        // K = koff .. koff+7
            af.q[1] = *(const uint4*)(ap + 16);   // K = koff+16 .. koff+23
            #pragma unroll
            for (int f = 0; f < 2; ++f)
                acc[sm][f].v = wmma_bf16(af, bf[f], acc[sm][f].v);
        }
    }

    // epilogue: C layout — VGPR v, lane L: row = v + 8*(L>>4), col = L&15
    #pragma unroll
    for (int sm = 0; sm < 4; ++sm) {
        const int row0 = mBase + mw * 64 + sm * 16 + hs * 8;
        #pragma unroll
        for (int f = 0; f < 2; ++f) {
            const int col = nBase + (nw * 2 + f) * 16 + l15;
            const float bv = bias[col];
            #pragma unroll
            for (int v = 0; v < 8; ++v) {
                float val = acc[sm][f].f[v] + bv;
                const size_t idx = (size_t)(row0 + v) * ldc + col;
                if (epi == 2) {
                    ((float*)outp)[idx] = val + resid[idx];
                } else {
                    if (epi == 1) val = 0.5f * val * (1.0f + erff(val * 0.70710678118f));
                    ((unsigned short*)outp)[idx] = f2bf(val);
                }
            }
        }
    }
}

// ---------------- flash attention (causal), bf16 WMMA ----------------
// qkv: bf16 [B, S, 3072] (Q | K | V interleaved per row), o: bf16 [B, S, 1024]
// grid: (S/128, NHEAD, BATCH), block 256 = 8 waves, wave owns 16 q rows.
__global__ __launch_bounds__(256) void attn_kernel(const unsigned short* __restrict__ qkv,
                                                   unsigned short* __restrict__ o) {
    __shared__ unsigned short Vt[64 * 72];        // V chunk transposed: [d][kpos]
    __shared__ unsigned short Pw[8 * 16 * 72];    // per-wave P tile [16 q][64 k]
    const int lane = threadIdx.x & 31;
    const int wid  = threadIdx.x >> 5;
    const int l15 = lane & 15, hs = lane >> 4;
    const int qblk = blockIdx.x, h = blockIdx.y, b = blockIdx.z;
    const int qbase = qblk * 128 + wid * 16;
    const size_t tokBase = (size_t)b * SEQ;

    // Q fragments (row-major, K hs*8..+7 and 16+hs*8..+7 per lane) — kept in registers
    Frag16 qf[2];
    {
        const unsigned short* qp = qkv + (tokBase + qbase + l15) * 3072 + h * HDIM;
        #pragma unroll
        for (int ks = 0; ks < 2; ++ks) {
            qf[ks].q[0] = *(const uint4*)(qp + ks * 32 + hs * 8);
            qf[ks].q[1] = *(const uint4*)(qp + ks * 32 + hs * 8 + 16);
        }
    }

    FragC accO[4];
    #pragma unroll
    for (int g = 0; g < 4; ++g)
        #pragma unroll
        for (int e = 0; e < 8; ++e) accO[g].f[e] = 0.f;
    float mrow[8], lrow[8];
    #pragma unroll
    for (int v = 0; v < 8; ++v) { mrow[v] = -1e30f; lrow[v] = 0.f; }

    const int nChunks = 2 * (qblk + 1);            // covers keys 0 .. qblk*128+127
    for (int kc = 0; kc < nChunks; ++kc) {
        const int kcbase = kc * 64;
        __syncthreads();
        {   // cooperative transposed V load: Vt[d][kpos]
            const int kp = threadIdx.x >> 2;
            const int dseg = (threadIdx.x & 3) * 16;
            const unsigned short* vp = qkv + (tokBase + kcbase + kp) * 3072 + 2048 + h * HDIM + dseg;
            Frag16 tmp;
            tmp.q[0] = ((const uint4*)vp)[0];
            tmp.q[1] = ((const uint4*)vp)[1];
            #pragma unroll
            for (int i = 0; i < 16; ++i) Vt[(dseg + i) * 72 + kp] = tmp.s[i];
        }
        __syncthreads();

        // scores: S = Q x K^T (K rows are K^T columns -> contiguous global loads)
        FragC sfr[4];
        #pragma unroll
        for (int f = 0; f < 4; ++f) {
            #pragma unroll
            for (int e = 0; e < 8; ++e) sfr[f].f[e] = 0.f;
            const unsigned short* kp_ = qkv + (tokBase + kcbase + f * 16 + l15) * 3072
                                        + 1024 + h * HDIM + hs * 8;
            #pragma unroll
            for (int ks = 0; ks < 2; ++ks) {
                Frag16 bf;
                bf.q[0] = *(const uint4*)(kp_ + ks * 32);
                bf.q[1] = *(const uint4*)(kp_ + ks * 32 + 16);
                sfr[f].v = wmma_bf16(qf[ks], bf, sfr[f].v);
            }
        }
        // scale + causal mask
        #pragma unroll
        for (int f = 0; f < 4; ++f) {
            const int kpos = kcbase + f * 16 + l15;
            #pragma unroll
            for (int v = 0; v < 8; ++v) {
                const int qpos = qbase + v + 8 * hs;
                float sv = sfr[f].f[v];
                sfr[f].f[v] = (kpos <= qpos) ? sv * 0.125f : -1e6f;
            }
        }
        // online softmax (rows live in 16-lane halves; xor masks < 16 stay inside)
        float alpha[8];
        #pragma unroll
        for (int v = 0; v < 8; ++v) {
            float loc = fmaxf(fmaxf(sfr[0].f[v], sfr[1].f[v]), fmaxf(sfr[2].f[v], sfr[3].f[v]));
            for (int off = 1; off < 16; off <<= 1) loc = fmaxf(loc, __shfl_xor(loc, off));
            const float mn = fmaxf(mrow[v], loc);
            alpha[v] = __expf(mrow[v] - mn);
            mrow[v] = mn;
            float ps = 0.f;
            #pragma unroll
            for (int f = 0; f < 4; ++f) {
                float p = __expf(sfr[f].f[v] - mn);
                sfr[f].f[v] = p;
                ps += p;
            }
            for (int off = 1; off < 16; off <<= 1) ps += __shfl_xor(ps, off);
            lrow[v] = lrow[v] * alpha[v] + ps;
        }
        #pragma unroll
        for (int g = 0; g < 4; ++g)
            #pragma unroll
            for (int v = 0; v < 8; ++v) accO[g].f[v] *= alpha[v];

        // P: C layout -> LDS -> A layout (per-wave private region, in-order LDS)
        unsigned short* P = &Pw[wid * 16 * 72];
        #pragma unroll
        for (int f = 0; f < 4; ++f)
            #pragma unroll
            for (int v = 0; v < 8; ++v)
                P[(v + 8 * hs) * 72 + f * 16 + l15] = f2bf(sfr[f].f[v]);

        // O += P x V  (V^T staged in LDS -> contiguous B fragments)
        #pragma unroll
        for (int ks2 = 0; ks2 < 2; ++ks2) {
            Frag16 af;
            const unsigned short* pp = &P[l15 * 72 + ks2 * 32 + hs * 8];
            af.q[0] = *(const uint4*)(pp);
            af.q[1] = *(const uint4*)(pp + 16);
            #pragma unroll
            for (int g = 0; g < 4; ++g) {
                Frag16 bf;
                const unsigned short* vp2 = &Vt[(g * 16 + l15) * 72 + ks2 * 32 + hs * 8];
                bf.q[0] = *(const uint4*)(vp2);
                bf.q[1] = *(const uint4*)(vp2 + 16);
                accO[g].v = wmma_bf16(af, bf, accO[g].v);
            }
        }
    }

    // normalize + store o (bf16, head-merged layout [B,S,D])
    #pragma unroll
    for (int g = 0; g < 4; ++g)
        #pragma unroll
        for (int v = 0; v < 8; ++v) {
            const int q = qbase + v + 8 * hs;
            o[(tokBase + q) * DMODEL + h * HDIM + g * 16 + l15] = f2bf(accO[g].f[v] / lrow[v]);
        }
}

// ---------------- host-side orchestration ----------------
extern "C" void kernel_launch(void* const* d_in, const int* in_sizes, int n_in,
                              void* d_out, int out_size, void* d_ws, size_t ws_size,
                              hipStream_t stream) {
    (void)in_sizes; (void)n_in; (void)out_size; (void)ws_size;
    const float* x     = (const float*)d_in[0];
    const float* ln1_g = (const float*)d_in[1];
    const float* ln1_b = (const float*)d_in[2];
    const float* wq    = (const float*)d_in[3];
    const float* bq    = (const float*)d_in[4];
    const float* wk    = (const float*)d_in[5];
    const float* bk    = (const float*)d_in[6];
    const float* wv    = (const float*)d_in[7];
    const float* bv    = (const float*)d_in[8];
    const float* wo    = (const float*)d_in[9];
    const float* bo    = (const float*)d_in[10];
    const float* ln2_g = (const float*)d_in[11];
    const float* ln2_b = (const float*)d_in[12];
    const float* w1    = (const float*)d_in[13];
    const float* b1    = (const float*)d_in[14];
    const float* w2    = (const float*)d_in[15];
    const float* b2    = (const float*)d_in[16];

    // bump allocator over d_ws (256B aligned chunks)
    char* ws = (char*)d_ws;
    size_t off = 0;
    auto alloc = [&](size_t bytes) -> void* {
        off = (off + 255) & ~(size_t)255;
        void* p = ws + off;
        off += bytes;
        return p;
    };
    unsigned short* h_bf   = (unsigned short*)alloc((size_t)NROWS * DMODEL * 2);
    unsigned short* qkv_bf = (unsigned short*)alloc((size_t)NROWS * 3072 * 2);
    unsigned short* o_bf   = (unsigned short*)alloc((size_t)NROWS * DMODEL * 2);
    float*          add1   = (float*)alloc((size_t)NROWS * DMODEL * 4);
    unsigned short* h2_bf  = (unsigned short*)alloc((size_t)NROWS * DMODEL * 2);
    unsigned short* mid_bf = (unsigned short*)alloc((size_t)NROWS * FFDIM * 2);
    unsigned* pQKV = (unsigned*)alloc((size_t)3 * DMODEL * DMODEL * 2);
    unsigned* pWO  = (unsigned*)alloc((size_t)DMODEL * DMODEL * 2);
    unsigned* pW1  = (unsigned*)alloc((size_t)DMODEL * FFDIM * 2);
    unsigned* pW2  = (unsigned*)alloc((size_t)FFDIM * DMODEL * 2);
    float* biasQKV = (float*)alloc(3072 * 4);

    const int dwDD = (DMODEL * DMODEL) >> 1;  // dwords per DxD packed weight
    const int dwDF = (DMODEL * FFDIM) >> 1;

    // 1) pack weights to bf16 fragment order + concat qkv bias
    pack_b_kernel<<<dwDD / 256, 256, 0, stream>>>(wq, pQKV,            DMODEL, DMODEL);
    pack_b_kernel<<<dwDD / 256, 256, 0, stream>>>(wk, pQKV + dwDD,     DMODEL, DMODEL);
    pack_b_kernel<<<dwDD / 256, 256, 0, stream>>>(wv, pQKV + 2 * dwDD, DMODEL, DMODEL);
    pack_b_kernel<<<dwDD / 256, 256, 0, stream>>>(wo, pWO,             DMODEL, DMODEL);
    pack_b_kernel<<<dwDF / 256, 256, 0, stream>>>(w1, pW1,             DMODEL, FFDIM);
    pack_b_kernel<<<dwDF / 256, 256, 0, stream>>>(w2, pW2,             FFDIM, DMODEL);
    concat3_kernel<<<12, 256, 0, stream>>>(bq, bk, bv, biasQKV);

    // 2) LN1: x -> h (bf16)
    ln_bf16_kernel<<<NROWS, 256, 0, stream>>>(x, ln1_g, ln1_b, h_bf);

    // 3) fused QKV GEMM: [4096,1024] x [1024,3072] -> qkv bf16
    gemm_bf16_wmma<<<dim3(3072 / 128, NROWS / 128), 256, 0, stream>>>(
        h_bf, DMODEL, pQKV, DMODEL, biasQKV, nullptr, qkv_bf, 3072, 0);

    // 4) causal flash attention -> o bf16
    attn_kernel<<<dim3(SEQ / 128, NHEAD, BATCH), 256, 0, stream>>>(qkv_bf, o_bf);

    // 5) output projection + residual: add1 = x + o@wo + bo (f32)
    gemm_bf16_wmma<<<dim3(DMODEL / 128, NROWS / 128), 256, 0, stream>>>(
        o_bf, DMODEL, pWO, DMODEL, bo, x, add1, DMODEL, 2);

    // 6) LN2: add1 -> h2 (bf16)
    ln_bf16_kernel<<<NROWS, 256, 0, stream>>>(add1, ln2_g, ln2_b, h2_bf);

    // 7) MLP up + exact gelu: mid = gelu(h2@w1 + b1) (bf16)
    gemm_bf16_wmma<<<dim3(FFDIM / 128, NROWS / 128), 256, 0, stream>>>(
        h2_bf, DMODEL, pW1, DMODEL, b1, nullptr, mid_bf, FFDIM, 1);

    // 8) MLP down + residual: out = add1 + mid@w2 + b2 (f32)
    gemm_bf16_wmma<<<dim3(DMODEL / 128, NROWS / 128), 256, 0, stream>>>(
        mid_bf, FFDIM, pW2, FFDIM, b2, add1, (float*)d_out, DMODEL, 2);
}
